// GraphDecoder_2714419331083
// MI455X (gfx1250) — compile-verified, matrix operand order checked by reading, not run on hardware
//
#include <hip/hip_runtime.h>
#include <cstdint>
#include <cstddef>

#define C_ 256
#define D_ 512
#define N_ 2048
#define B_ 32
#define E_ 32768

typedef __attribute__((ext_vector_type(16))) __bf16 v16bf;
typedef __attribute__((ext_vector_type(8)))  float  v8f;

__device__ __forceinline__ unsigned short f2bf(float f) {
  unsigned u = __float_as_uint(f);
  u += 0x7FFFu + ((u >> 16) & 1u);           // round-to-nearest-even
  return (unsigned short)(u >> 16);
}
__device__ __forceinline__ float bf2f(unsigned short h) {
  return __uint_as_float(((unsigned)h) << 16);
}

// LDS byte offset = low 32 bits of the flat shared-aperture address (ISA 10.2)
__device__ __forceinline__ unsigned lds_off32(const void* p) {
  return (unsigned)(uintptr_t)p;
}
// async global->LDS 16B copy per lane; tracked by ASYNCcnt
__device__ __forceinline__ void async_ld_b128(unsigned lds, const void* g) {
  asm volatile("global_load_async_to_lds_b128 %0, %1, off"
               :: "v"(lds), "v"((unsigned long long)(uintptr_t)g)
               : "memory");
}
__device__ __forceinline__ void wait_async0() {
  asm volatile("s_wait_asynccnt 0x0" ::: "memory");
}

// ---------------- weight fp32 -> bf16 ----------------
__global__ void cvt_w_kernel(const float* __restrict__ src,
                             unsigned short* __restrict__ dst, int n) {
  for (int i = blockIdx.x * blockDim.x + threadIdx.x; i < n;
       i += gridDim.x * blockDim.x)
    dst[i] = f2bf(src[i]);
}

// ---------------- stage A: fc + FiLM (gamma/beta per batch) ----------------
__global__ __launch_bounds__(512) void stage_a_kernel(
    const float* __restrict__ x, const float* __restrict__ fc_w,
    const float* __restrict__ fc_b, const float* __restrict__ fw1,
    const float* __restrict__ fb1, const float* __restrict__ fw2,
    const float* __restrict__ fb2, float* __restrict__ gamma,
    float* __restrict__ beta) {
  __shared__ float h0[D_];
  __shared__ float t[C_];
  const int b = blockIdx.x, j = threadIdx.x;
  float acc = fc_b[j];
  for (int c = 0; c < C_; ++c) acc += x[b * C_ + c] * fc_w[j * C_ + c];
  h0[j] = acc;
  __syncthreads();
  if (j < C_) {
    float a2 = fb1[j];
    for (int c = 0; c < D_; ++c) a2 += fw1[j * D_ + c] * h0[c];
    t[j] = fmaxf(a2, 0.f);
  }
  __syncthreads();
  for (int o = j; o < 2 * D_; o += D_) {
    float a3 = fb2[o];
    for (int c = 0; c < C_; ++c) a3 += fw2[o * C_ + c] * t[c];
    if (o < D_) gamma[b * D_ + o] = a3;
    else        beta[b * D_ + (o - D_)] = a3;
  }
}

// ---------------- stage B: H1[b][n][c] = bf16(gamma*pe + beta) -------------
__global__ __launch_bounds__(256) void stage_b_kernel(
    const float* __restrict__ gamma, const float* __restrict__ beta,
    unsigned short* __restrict__ H1) {
  const int n = blockIdx.x, b = blockIdx.y;
  const float kdiv = -5.298317367f / 512.0f;  // -ln(200)/d
  const size_t base = ((size_t)(b * N_ + n)) * D_;
  for (int c = threadIdx.x; c < D_; c += blockDim.x) {
    float div = __expf((float)(c & ~1) * kdiv);
    float ang = (float)n * div;
    float pe  = (c & 1) ? __cosf(ang) : __sinf(ang);
    H1[base + c] = f2bf(gamma[b * D_ + c] * pe + beta[b * D_ + c]);
  }
}

// ---------------- WMMA GEMM + bias + optional BN stats ---------------------
// Y[b,n,m] = sum_k W[m,k] * H[b,n,k] + bias[m]
// H node-major (B,N_,K) bf16.  Macro tile 128(M)x128(N), K-step 32,
// double-buffered LDS fed by global_load_async_to_lds_b128 (ASYNCcnt).
__global__ __launch_bounds__(256) void gemm_bn_kernel(
    const unsigned short* __restrict__ Hb,   // (B, N_, K) bf16
    const unsigned short* __restrict__ Wb,   // (M, K) bf16
    const float* __restrict__ bias,          // (M)
    float* __restrict__ Yout,                // (B, N_, M) fp32 (if Obf==null)
    unsigned short* __restrict__ Obf,        // (B, N_, M) bf16 (optional)
    float* __restrict__ gsum, float* __restrict__ gsq,  // (M) stats or null
    int M, int K) {
  __shared__ unsigned short As[2][128][40];  // pad 8: conflict-free frags
  __shared__ unsigned short Bs[2][128][40];
  __shared__ float sSum[128];
  __shared__ float sSq[128];

  const int tid  = threadIdx.x;
  const int lane = tid & 31, wave = tid >> 5;
  const int wm = wave & 3, wn = wave >> 2;     // 4 waves in M, 2 in N
  const int lm = lane & 15, kh = lane >> 4;
  const int blockN = blockIdx.x * 128;
  const int blockM = blockIdx.y * 128;
  const int b = blockIdx.z;

  if (gsum && tid < 128) { sSum[tid] = 0.f; sSq[tid] = 0.f; }

  v8f acc[2][4];
#pragma unroll
  for (int i = 0; i < 2; i++)
#pragma unroll
    for (int j = 0; j < 4; j++)
#pragma unroll
      for (int r = 0; r < 8; r++) acc[i][j][r] = 0.f;

  // Cooperative tile load map: 256 threads, each owns 32B (16 bf16) of a row
  const int ldr = tid >> 1, ldk = (tid & 1) << 4;  // row 0..127, k 0|16
  const unsigned short* wrow = Wb + (size_t)(blockM + ldr) * K + ldk;
  const unsigned short* hrow = Hb + ((size_t)b * N_ + blockN + ldr) * K + ldk;
  unsigned aoff[2], boff[2];
  aoff[0] = lds_off32(&As[0][ldr][ldk]);
  aoff[1] = lds_off32(&As[1][ldr][ldk]);
  boff[0] = lds_off32(&Bs[0][ldr][ldk]);
  boff[1] = lds_off32(&Bs[1][ldr][ldk]);

  union Frag { v16bf v; unsigned u[8]; };

  // preload K-tile 0 into buffer 0
  async_ld_b128(aoff[0],      wrow);
  async_ld_b128(aoff[0] + 16, wrow + 8);
  async_ld_b128(boff[0],      hrow);
  async_ld_b128(boff[0] + 16, hrow + 8);

  const int nk = K >> 5;
  for (int kt = 0; kt < nk; ++kt) {
    const int cur = kt & 1, nxt = cur ^ 1;
    wait_async0();          // my contributions to buf 'cur' landed
    __syncthreads();        // everyone's landed; buf 'nxt' free to overwrite

    if (kt + 1 < nk) {      // overlap next tile's DMA with this tile's WMMA
      const unsigned short* wnx = wrow + (size_t)(kt + 1) * 32;
      const unsigned short* hnx = hrow + (size_t)(kt + 1) * 32;
      async_ld_b128(aoff[nxt],      wnx);
      async_ld_b128(aoff[nxt] + 16, wnx + 8);
      async_ld_b128(boff[nxt],      hnx);
      async_ld_b128(boff[nxt] + 16, hnx + 8);
    }

    // Fragment gather per CDNA5 16-bit A/B layout: dword v<4 -> K-pair v+4kh,
    // v>=4 -> K-pair v+4+4kh (contiguous dwords -> ds_load_b128 pairs)
    Frag af[2], bfr[4];
#pragma unroll
    for (int i = 0; i < 2; i++) {
      const unsigned* arow = (const unsigned*)&As[cur][wm * 32 + i * 16 + lm][0];
#pragma unroll
      for (int v = 0; v < 8; v++)
        af[i].u[v] = arow[((v < 4) ? v : v + 4) + 4 * kh];
    }
#pragma unroll
    for (int j = 0; j < 4; j++) {
      const unsigned* brow = (const unsigned*)&Bs[cur][wn * 64 + j * 16 + lm][0];
#pragma unroll
      for (int v = 0; v < 8; v++)
        bfr[j].u[v] = brow[((v < 4) ? v : v + 4) + 4 * kh];
    }
#pragma unroll
    for (int i = 0; i < 2; i++)
#pragma unroll
      for (int j = 0; j < 4; j++)
        acc[i][j] = __builtin_amdgcn_wmma_f32_16x16x32_bf16(
            false, af[i].v, false, bfr[j].v, (short)0, acc[i][j], false, false);
  }

  // Epilogue: bias, node-major stores, BN partial stats
#pragma unroll
  for (int i = 0; i < 2; i++) {
    const int mbase = wm * 32 + i * 16 + 8 * kh;  // local channel base
    float sloc[8], qloc[8];
#pragma unroll
    for (int r = 0; r < 8; r++) { sloc[r] = 0.f; qloc[r] = 0.f; }
#pragma unroll
    for (int j = 0; j < 4; j++) {
      const int node = blockN + wn * 64 + j * 16 + lm;
      const size_t obase = ((size_t)b * N_ + node) * M + blockM + mbase;
      float v[8];
#pragma unroll
      for (int r = 0; r < 8; r++) {
        float xv = acc[i][j][r] + bias[blockM + mbase + r];
        v[r] = xv;
        sloc[r] += xv;
        qloc[r] += xv * xv;
      }
      if (Obf) {
        uint4 pk;
        pk.x = (unsigned)f2bf(v[0]) | ((unsigned)f2bf(v[1]) << 16);
        pk.y = (unsigned)f2bf(v[2]) | ((unsigned)f2bf(v[3]) << 16);
        pk.z = (unsigned)f2bf(v[4]) | ((unsigned)f2bf(v[5]) << 16);
        pk.w = (unsigned)f2bf(v[6]) | ((unsigned)f2bf(v[7]) << 16);
        *(uint4*)&Obf[obase] = pk;
      } else {
        *(float4*)&Yout[obase]     = make_float4(v[0], v[1], v[2], v[3]);
        *(float4*)&Yout[obase + 4] = make_float4(v[4], v[5], v[6], v[7]);
      }
    }
    if (gsum) {
#pragma unroll
      for (int r = 0; r < 8; r++) {
        float s = sloc[r], q = qloc[r];
#pragma unroll
        for (int msk = 1; msk < 16; msk <<= 1) {
          s += __shfl_xor(s, msk, 32);
          q += __shfl_xor(q, msk, 32);
        }
        if (lm == 0) {
          atomicAdd(&sSum[mbase + r], s);   // ds_add_f32
          atomicAdd(&sSq[mbase + r], q);
        }
      }
    }
  }
  if (gsum) {
    __syncthreads();
    if (tid < 128) {
      unsafeAtomicAdd(&gsum[blockM + tid], sSum[tid]);
      unsafeAtomicAdd(&gsq[blockM + tid], sSq[tid]);
    }
  }
}

// ---------------- BN finalize: per-channel scale/shift ---------------------
__global__ void bn_finalize_kernel(const float* __restrict__ gsum,
                                   const float* __restrict__ gsq,
                                   const float* __restrict__ g,
                                   const float* __restrict__ be,
                                   float* __restrict__ scale,
                                   float* __restrict__ shift, int M) {
  int t = blockIdx.x * blockDim.x + threadIdx.x;
  if (t < M) {
    const float inv_n = 1.f / (float)(B_ * N_);
    float mean = gsum[t] * inv_n;
    float var  = gsq[t] * inv_n - mean * mean;
    float inv  = rsqrtf(var + 1e-5f);
    float sc   = g[t] * inv;
    scale[t]   = sc;
    shift[t]   = be[t] - mean * sc;
  }
}

// ---------------- BN apply + ReLU + cvt to bf16 ----------------------------
__global__ void bn_apply_kernel(const float* __restrict__ Y,
                                unsigned short* __restrict__ H,
                                const float* __restrict__ scale,
                                const float* __restrict__ shift, int M,
                                size_t total) {
  for (size_t i = (size_t)blockIdx.x * blockDim.x + threadIdx.x; i < total;
       i += (size_t)gridDim.x * blockDim.x) {
    int c = (int)(i % (size_t)M);
    float v = fmaxf(scale[c] * Y[i] + shift[c], 0.f);
    H[i] = f2bf(v);
  }
}

// ---------------- edge scatter (segment_sum), node-major -------------------
__global__ __launch_bounds__(256) void scatter_kernel(
    const unsigned short* __restrict__ H4, const long long* __restrict__ gl,
    float* __restrict__ agg) {
  const int e = blockIdx.x;
  const int src = (int)gl[e];
  const int dst = (int)gl[E_ + e];
  for (int b = 0; b < B_; ++b) {
    const size_t sb = ((size_t)b * N_ + src) * D_;
    const size_t db = ((size_t)b * N_ + dst) * D_;
    for (int c = threadIdx.x; c < D_; c += 256)
      unsafeAtomicAdd(&agg[db + c], bf2f(H4[sb + c]));  // global_atomic_add_f32
  }
}

// ---------------- fp32 -> bf16 bulk convert --------------------------------
__global__ void to_bf16_kernel(const float* __restrict__ src,
                               unsigned short* __restrict__ dst, size_t n) {
  for (size_t i = (size_t)blockIdx.x * blockDim.x + threadIdx.x; i < n;
       i += (size_t)gridDim.x * blockDim.x)
    dst[i] = f2bf(src[i]);
}

// ---------------- final 3-channel projection -------------------------------
__global__ __launch_bounds__(128) void final_kernel(
    const unsigned short* __restrict__ H6, const float* __restrict__ w2,
    const float* __restrict__ b2, float* __restrict__ out) {
  int idx = blockIdx.x * blockDim.x + threadIdx.x;  // b*N_+n
  if (idx >= B_ * N_) return;
  int b = idx / N_, n = idx % N_;
  const unsigned short* h = H6 + (size_t)idx * C_;
  float s0 = b2[0], s1 = b2[1], s2 = b2[2];
  for (int c = 0; c < C_; ++c) {
    float hv = bf2f(h[c]);
    s0 += hv * w2[c];
    s1 += hv * w2[C_ + c];
    s2 += hv * w2[2 * C_ + c];
  }
  out[((size_t)b * 3 + 0) * N_ + n] = s0;
  out[((size_t)b * 3 + 1) * N_ + n] = s1;
  out[((size_t)b * 3 + 2) * N_ + n] = s2;
}

extern "C" void kernel_launch(void* const* d_in, const int* in_sizes, int n_in,
                              void* d_out, int out_size, void* d_ws,
                              size_t ws_size, hipStream_t stream) {
  const float*     x      = (const float*)d_in[0];
  const long long* gl     = (const long long*)d_in[1];  // int64 indices
  const float* fc_w   = (const float*)d_in[2];
  const float* fc_b   = (const float*)d_in[3];
  const float* fw1    = (const float*)d_in[4];
  const float* fb1    = (const float*)d_in[5];
  const float* fw2    = (const float*)d_in[6];
  const float* fb2    = (const float*)d_in[7];
  const float* gc_w   = (const float*)d_in[8];
  const float* gc_b   = (const float*)d_in[9];
  const float* m2_w1  = (const float*)d_in[10];
  const float* m2_b1  = (const float*)d_in[11];
  const float* m2_g1  = (const float*)d_in[12];
  const float* m2_be1 = (const float*)d_in[13];
  const float* m2_w2  = (const float*)d_in[14];
  const float* m2_b2  = (const float*)d_in[15];
  const float* m1_w[3]  = {(const float*)d_in[16], (const float*)d_in[20],
                           (const float*)d_in[24]};
  const float* m1_b[3]  = {(const float*)d_in[17], (const float*)d_in[21],
                           (const float*)d_in[25]};
  const float* m1_g[3]  = {(const float*)d_in[18], (const float*)d_in[22],
                           (const float*)d_in[26]};
  const float* m1_be[3] = {(const float*)d_in[19], (const float*)d_in[23],
                           (const float*)d_in[27]};

  // ---- workspace layout ----
  char* ws = (char*)d_ws;
  unsigned short* wW[3];
  wW[0] = (unsigned short*)ws;
  wW[1] = wW[0] + D_ * D_;
  wW[2] = wW[1] + D_ * D_;
  unsigned short* wGc = wW[2] + D_ * D_;       // 256x512
  unsigned short* wM2 = wGc + C_ * D_;         // 256x256
  float* gamma = (float*)(wM2 + C_ * C_);      // 32x512
  float* beta  = gamma + B_ * D_;
  float* gsum  = beta + B_ * D_;               // 512 (gsq contiguous after)
  float* gsq   = gsum + D_;
  float* scale = gsq + D_;
  float* shift = scale + D_;
  size_t small_end = (size_t)((char*)(shift + D_) - ws);
  size_t big_off   = (small_end + 255) & ~(size_t)255;
  unsigned short* bufA = (unsigned short*)(ws + big_off);        // 64 MB bf16
  unsigned short* bufB = bufA + (size_t)B_ * N_ * D_;            // 64 MB bf16
  float*          bufY = (float*)(bufB + (size_t)B_ * N_ * D_);  // 134 MB f32

  const size_t totD = (size_t)B_ * N_ * D_;
  const size_t totC = (size_t)B_ * N_ * C_;

  // ---- weights to bf16 ----
  cvt_w_kernel<<<256, 256, 0, stream>>>(m1_w[0], wW[0], D_ * D_);
  cvt_w_kernel<<<256, 256, 0, stream>>>(m1_w[1], wW[1], D_ * D_);
  cvt_w_kernel<<<256, 256, 0, stream>>>(m1_w[2], wW[2], D_ * D_);
  cvt_w_kernel<<<256, 256, 0, stream>>>(gc_w, wGc, C_ * D_);
  cvt_w_kernel<<<256, 256, 0, stream>>>(m2_w1, wM2, C_ * C_);

  // ---- fc + FiLM, then gamma*PE+beta ----
  stage_a_kernel<<<B_, 512, 0, stream>>>(x, fc_w, fc_b, fw1, fb1, fw2, fb2,
                                         gamma, beta);
  stage_b_kernel<<<dim3(N_, B_), 256, 0, stream>>>(gamma, beta, bufA);

  // ---- three 512x512 WMMA GEMM + BN + ReLU layers (ping-pong A/B) ----
  dim3 ggridD(N_ / 128, D_ / 128, B_);
  unsigned short* hin  = bufA;
  unsigned short* hout = bufB;
  for (int L = 0; L < 3; ++L) {
    hipMemsetAsync(gsum, 0, 2 * D_ * sizeof(float), stream);
    gemm_bn_kernel<<<ggridD, 256, 0, stream>>>(hin, wW[L], m1_b[L], bufY,
                                               nullptr, gsum, gsq, D_, D_);
    bn_finalize_kernel<<<1, D_, 0, stream>>>(gsum, gsq, m1_g[L], m1_be[L],
                                             scale, shift, D_);
    bn_apply_kernel<<<2048, 256, 0, stream>>>(bufY, hout, scale, shift, D_,
                                              totD);
    unsigned short* t = hin; hin = hout; hout = t;
  }
  // after loop: H4 = hin (= bufB)

  // ---- graph scatter: agg[b, dst, :] += H4[b, src, :] ----
  hipMemsetAsync(bufY, 0, totD * sizeof(float), stream);
  scatter_kernel<<<E_, 256, 0, stream>>>(hin, gl, bufY);
  to_bf16_kernel<<<2048, 256, 0, stream>>>(bufY, hout, totD);  // hout = bufA

  // ---- gc conv: 256x512, bias only, direct bf16 out ----
  dim3 ggridC(N_ / 128, C_ / 128, B_);
  gemm_bn_kernel<<<ggridC, 256, 0, stream>>>(hout, wGc, gc_b, nullptr, hin,
                                             nullptr, nullptr, C_, D_);

  // ---- m2 layer: 256x256 GEMM + BN + ReLU ----
  hipMemsetAsync(gsum, 0, 2 * C_ * sizeof(float), stream);
  gemm_bn_kernel<<<ggridC, 256, 0, stream>>>(hin, wM2, m2_b1, bufY, nullptr,
                                             gsum, gsq, C_, C_);
  bn_finalize_kernel<<<1, C_, 0, stream>>>(gsum, gsq, m2_g1, m2_be1, scale,
                                           shift, C_);
  bn_apply_kernel<<<2048, 256, 0, stream>>>(bufY, hout, scale, shift, C_, totC);

  // ---- final 3-channel projection to d_out (B,3,N) fp32 ----
  final_kernel<<<(B_ * N_) / 128, 128, 0, stream>>>(hout, m2_w2, m2_b2,
                                                    (float*)d_out);
}
